// CVAE_CI3PP_Both_65970697666907
// MI455X (gfx1250) — matrix operand order repaired and checked by reading, not compiled, not run on hardware
//
#include <hip/hip_runtime.h>
#include <cstdint>
#include <cstddef>

// ---------------------------------------------------------------------------
// CVAE CI3PP forward for MI455X (gfx1250, wave32).
// - 5 GRUs: v_wmma_f32_16x16x32_bf16, 16 batch rows/wave, LDS-staged weights,
//   dual accumulator chains, padded LDS strides (rounds 1-3).
// - MHA QKV + output projections now also WMMA (token-major 16-row tiles);
//   only the head-dim-8 attention core remains VALU.
// ---------------------------------------------------------------------------

#define B_SZ   2048
#define T_OBS  60
#define T_PRED 80
#define RT_OBS ((B_SZ * T_OBS) / 16)   // 7680 token tiles

typedef __attribute__((ext_vector_type(16))) __bf16 v16bf;
typedef __attribute__((ext_vector_type(8)))  float  v8f;
typedef __attribute__((ext_vector_type(4)))  unsigned int v4u;

union BFrag { v16bf v; unsigned short u[16]; };

__device__ inline unsigned short f2bf(float f) {
  unsigned int x = __float_as_uint(f);
  x += 0x7FFFu + ((x >> 16) & 1u);   // round-to-nearest-even
  return (unsigned short)(x >> 16);
}

// A-fragment (16x32 bf16) element mapping, CDNA5 ISA 7.12.2:
//   lane = m + 16*hi ;  element j -> K = (j&7) + 8*hi + 16*(j>>3)
__device__ inline size_t afrag_idx(size_t tile_base, int m, int kk) {
  int grp  = kk >> 4;
  int hi   = (kk >> 3) & 1;
  int j    = (kk & 7) + 8 * grp;
  int lane = m + 16 * hi;
  return tile_base + (size_t)lane * 16 + j;
}

// ---------------------------------------------------------------------------
// Prepack weight W[N,K] (row-major, acts as B = W^T per 32x16 tile) into
// fragment order: dst[((nt*KT+kt)*32 + lane)*16 + j] = bf16(W[n][k])
// ---------------------------------------------------------------------------
__global__ void pack_w_kernel(const float* __restrict__ W, int N, int K,
                              unsigned short* __restrict__ dst) {
  int ktiles = K >> 5;
  int total  = (N >> 4) * ktiles * 512;
  int i = blockIdx.x * blockDim.x + threadIdx.x;
  if (i >= total) return;
  int j    = i & 15;
  int lane = (i >> 4) & 31;
  int tile = i >> 9;
  int kt   = tile % ktiles;
  int nt   = tile / ktiles;
  int n = nt * 16 + (lane & 15);
  int k = kt * 32 + (j & 7) + 8 * (lane >> 4) + 16 * (j >> 3);
  dst[i] = f2bf(W[(size_t)n * K + k]);
}

// ---------------------------------------------------------------------------
// Embedding -> token-major A-fragment bf16 (rows r = b*60+t, K=32 exact).
// ---------------------------------------------------------------------------
__global__ void emb_frag_kernel(const float* __restrict__ x,
                                const float* __restrict__ w,
                                const float* __restrict__ bias,
                                unsigned short* __restrict__ e_frag, int total) {
  int idx = blockIdx.x * blockDim.x + threadIdx.x;
  if (idx >= total) return;
  int e = idx & 31;
  int r = idx >> 5;                       // global token row
  float v = x[(size_t)r * 2 + 0] * w[e * 2 + 0] +
            x[(size_t)r * 2 + 1] * w[e * 2 + 1] + bias[e];
  v = fmaxf(v, 0.0f);
  e_frag[afrag_idx((size_t)(r >> 4) * 512, r & 15, e)] = f2bf(v);
}

// ---------------------------------------------------------------------------
// QKV projection GEMM (WMMA): one wave per 16-token tile, N=96 (Q|K|V),
// K=32 exact. p=0 uses the q-source fragments, p=1,2 the kv-source.
// Output row-major f32 [B*60, 96].
// ---------------------------------------------------------------------------
__global__ __launch_bounds__(256)
void qkv_gemm_kernel(const unsigned short* __restrict__ qf,
                     const unsigned short* __restrict__ kvf,
                     const unsigned short* __restrict__ inw_frag, // [6][512]
                     const float* __restrict__ in_b,              // [96]
                     float* __restrict__ qkv) {
  const int lane = threadIdx.x & 31;
  const int wv   = threadIdx.x >> 5;
  const int rt   = blockIdx.x * 8 + wv;
  const int m    = lane & 15;
  const int hi   = lane >> 4;
  v16bf aq  = *reinterpret_cast<const v16bf*>(qf  + (size_t)rt * 512 + lane * 16);
  v16bf akv = *reinterpret_cast<const v16bf*>(kvf + (size_t)rt * 512 + lane * 16);
  for (int p = 0; p < 3; ++p) {
    v16bf a = (p == 0) ? aq : akv;
    v8f acc0, acc1;
    float bv0 = in_b[p * 32 + m];
    float bv1 = in_b[p * 32 + 16 + m];
#pragma unroll
    for (int r = 0; r < 8; ++r) { acc0[r] = bv0; acc1[r] = bv1; }
    v16bf w0 = *reinterpret_cast<const v16bf*>(inw_frag + (size_t)(p * 2 + 0) * 512 + lane * 16);
    v16bf w1 = *reinterpret_cast<const v16bf*>(inw_frag + (size_t)(p * 2 + 1) * 512 + lane * 16);
    acc0 = __builtin_amdgcn_wmma_f32_16x16x32_bf16(false, a, false, w0, (short)0, acc0, false, false);
    acc1 = __builtin_amdgcn_wmma_f32_16x16x32_bf16(false, a, false, w1, (short)0, acc1, false, false);
#pragma unroll
    for (int r = 0; r < 8; ++r) {
      size_t row = (size_t)rt * 16 + r + 8 * hi;
      qkv[row * 96 + p * 32 + m]      = acc0[r];
      qkv[row * 96 + p * 32 + 16 + m] = acc1[r];
    }
  }
}

// ---------------------------------------------------------------------------
// Attention core (VALU; head dim 8 does not map onto 16x16x32 tiles).
// One block (64 thr) per batch element. Emits O in token-major A-frag bf16.
// ---------------------------------------------------------------------------
__global__ __launch_bounds__(64)
void attn_kernel(const float* __restrict__ qkv,
                 unsigned short* __restrict__ o_frag) {
  const int b = blockIdx.x, tid = threadIdx.x;
  extern __shared__ float sm[];
  float* Q  = sm;          // 60*32
  float* Kd = Q + 1920;    // 60*32
  float* V  = Kd + 1920;   // 60*32
  float* S  = V + 1920;    // 60*60 (reused per head)
  float* O  = S + 3600;    // 60*32

  for (int idx = tid; idx < T_OBS * 32; idx += 64) {
    int t = idx >> 5, e = idx & 31;
    size_t row = (size_t)b * T_OBS + t;
    Q[idx]  = qkv[row * 96 + e];
    Kd[idx] = qkv[row * 96 + 32 + e];
    V[idx]  = qkv[row * 96 + 64 + e];
  }
  __syncthreads();

  const float scale = 0.35355339059327373f;  // 1/sqrt(8)
  for (int h = 0; h < 4; ++h) {
    for (int idx = tid; idx < T_OBS * T_OBS; idx += 64) {
      int tq = idx / T_OBS, tk = idx - tq * T_OBS;
      float s = 0.0f;
      for (int d = 0; d < 8; ++d)
        s += Q[tq * 32 + h * 8 + d] * Kd[tk * 32 + h * 8 + d];
      S[idx] = s * scale;
    }
    __syncthreads();
    if (tid < T_OBS) {
      float mx = -1e30f;
      for (int tk = 0; tk < T_OBS; ++tk) mx = fmaxf(mx, S[tid * T_OBS + tk]);
      float sum = 0.0f;
      for (int tk = 0; tk < T_OBS; ++tk) {
        float e2 = __expf(S[tid * T_OBS + tk] - mx);
        S[tid * T_OBS + tk] = e2; sum += e2;
      }
      float inv = 1.0f / sum;
      for (int tk = 0; tk < T_OBS; ++tk) S[tid * T_OBS + tk] *= inv;
    }
    __syncthreads();
    for (int idx = tid; idx < T_OBS * 8; idx += 64) {
      int t = idx >> 3, d = idx & 7;
      float s = 0.0f;
      for (int tk = 0; tk < T_OBS; ++tk)
        s += S[t * T_OBS + tk] * V[tk * 32 + h * 8 + d];
      O[t * 32 + h * 8 + d] = s;
    }
    __syncthreads();
  }

  for (int idx = tid; idx < T_OBS * 32; idx += 64) {
    int t = idx >> 5, e = idx & 31;
    int r = b * T_OBS + t;
    o_frag[afrag_idx((size_t)(r >> 4) * 512, r & 15, e)] = f2bf(O[t * 32 + e]);
  }
}

// ---------------------------------------------------------------------------
// Output projection GEMM (WMMA): [tokens,32] x [32,32], one wave per 16-token
// tile, dual chains; scatters into the GRU concat fragment layout (bf16).
// ---------------------------------------------------------------------------
__global__ __launch_bounds__(256)
void outproj_gemm_kernel(const unsigned short* __restrict__ o_frag,
                         const unsigned short* __restrict__ ow_frag, // [2][512]
                         const float* __restrict__ out_b,            // [32]
                         unsigned short* __restrict__ dst, int dst_kt) {
  const int lane = threadIdx.x & 31;
  const int wv   = threadIdx.x >> 5;
  const int rt   = blockIdx.x * 8 + wv;
  const int m    = lane & 15;
  const int hi   = lane >> 4;
  v16bf a  = *reinterpret_cast<const v16bf*>(o_frag + (size_t)rt * 512 + lane * 16);
  v16bf w0 = *reinterpret_cast<const v16bf*>(ow_frag + (size_t)0 * 512 + lane * 16);
  v16bf w1 = *reinterpret_cast<const v16bf*>(ow_frag + (size_t)1 * 512 + lane * 16);
  v8f acc0, acc1;
  float bv0 = out_b[m], bv1 = out_b[16 + m];
#pragma unroll
  for (int r = 0; r < 8; ++r) { acc0[r] = bv0; acc1[r] = bv1; }
  acc0 = __builtin_amdgcn_wmma_f32_16x16x32_bf16(false, a, false, w0, (short)0, acc0, false, false);
  acc1 = __builtin_amdgcn_wmma_f32_16x16x32_bf16(false, a, false, w1, (short)0, acc1, false, false);
#pragma unroll
  for (int r = 0; r < 8; ++r) {
    int row = rt * 16 + r + 8 * hi;
    int bb = row / T_OBS, tt = row - bb * T_OBS;
    size_t base = (((size_t)(bb >> 4) * T_OBS + tt) * 2 + dst_kt) * 512;
    dst[afrag_idx(base, bb & 15, m)]      = f2bf(acc0[r]);
    dst[afrag_idx(base, bb & 15, 16 + m)] = f2bf(acc1[r]);
  }
}

// ---------------------------------------------------------------------------
// y_lin -> A-fragment bf16 for the y-GRU (KIN=128 -> 4 k-tiles).
// ---------------------------------------------------------------------------
__global__ void ylin_kernel(const float* __restrict__ y,
                            const float* __restrict__ w,
                            const float* __restrict__ bias,
                            unsigned short* __restrict__ y_frag, int total) {
  int idx = blockIdx.x * blockDim.x + threadIdx.x;
  if (idx >= total) return;
  int f  = idx & 127;
  int bt = idx >> 7;
  int b  = bt / T_PRED;
  int t  = bt - b * T_PRED;
  float v = tanhf(y[(size_t)bt * 2 + 0] * w[f * 2 + 0] +
                  y[(size_t)bt * 2 + 1] * w[f * 2 + 1] + bias[f]);
  int btile = b >> 4, m = b & 15, kt = f >> 5, kk = f & 31;
  size_t base = ((((size_t)btile * T_PRED + t) * 4 + kt)) * 512;
  y_frag[afrag_idx(base, m, kk)] = f2bf(v);
}

// ---------------------------------------------------------------------------
// Latent + decoder input. One block (64 thr) per batch element.
// ---------------------------------------------------------------------------
__global__ __launch_bounds__(64)
void latent_kernel(const float* __restrict__ ht, const float* __restrict__ hc,
                   const float* __restrict__ ha, const float* __restrict__ hy,
                   const float* __restrict__ eps,
                   const float* __restrict__ mu_w, const float* __restrict__ mu_b,
                   const float* __restrict__ var_w, const float* __restrict__ var_b,
                   const float* __restrict__ dec_w, const float* __restrict__ dec_b,
                   float* __restrict__ mean_out, float* __restrict__ lv_out,
                   unsigned short* __restrict__ dl_frag) {
  int b = blockIdx.x, tid = threadIdx.x;
  __shared__ float cat[640];
  __shared__ float zmean[32], zlv[32];
  __shared__ float dec_in[416];
  for (int i = tid; i < 128; i += 64) {
    cat[i]       = ht[(size_t)b * 128 + i];
    cat[128 + i] = hc[(size_t)b * 128 + i];
    cat[256 + i] = ha[(size_t)b * 128 + i];
  }
  for (int i = tid; i < 256; i += 64) cat[384 + i] = hy[(size_t)b * 256 + i];
  __syncthreads();
  for (int i = tid; i < 384; i += 64) dec_in[i] = cat[i];
  if (tid < 32) {
    float s = mu_b[tid];
    for (int k = 0; k < 640; ++k) s += cat[k] * mu_w[tid * 640 + k];
    zmean[tid] = s;
    mean_out[(size_t)b * 32 + tid] = s;
  } else {
    int e = tid - 32;
    float s = var_b[e];
    for (int k = 0; k < 640; ++k) s += cat[k] * var_w[e * 640 + k];
    zlv[e] = s;
    lv_out[(size_t)b * 32 + e] = s;
  }
  __syncthreads();
  if (tid < 32)
    dec_in[384 + tid] = eps[(size_t)b * 32 + tid] * __expf(0.5f * zlv[tid]) + zmean[tid];
  __syncthreads();
  int btile = b >> 4, m = b & 15;
  for (int f = tid; f < 128; f += 64) {
    float s = dec_b[f];
    for (int k = 0; k < 416; ++k) s += dec_in[k] * dec_w[f * 416 + k];
    s = fmaxf(s, 0.0f);
    int kt = f >> 5, kk = f & 31;
    size_t base = ((size_t)btile * 4 + kt) * 512;
    dl_frag[afrag_idx(base, m, kk)] = f2bf(s);
  }
}

// ---------------------------------------------------------------------------
// WMMA GRU. 2 waves per workgroup; each wave owns 16 batch rows; weights
// optionally staged once into LDS and shared by both waves.
// Dual accumulator chains per rolled nt iteration hide WMMA RAW hazards.
// Row strides padded: hbuf H+2 words, gi/gh 3H+2 words (bank spread).
// ---------------------------------------------------------------------------
template <int KIN, int H, int T, bool XCONST, bool HEADOUT, bool STAGE_W>
__global__ __launch_bounds__(64)
void gru_wmma_kernel(const unsigned short* __restrict__ x_frag,
                     const unsigned short* __restrict__ wih_frag,
                     const unsigned short* __restrict__ whh_frag,
                     const float* __restrict__ bih, const float* __restrict__ bhh,
                     float* __restrict__ h_out,
                     const float* __restrict__ head_w,
                     const float* __restrict__ head_b,
                     float* __restrict__ pred) {
  constexpr int KTX = KIN / 32;
  constexpr int KTH = H / 32;
  constexpr int NT  = (3 * H) / 16;
  constexpr int TX  = XCONST ? 1 : T;
  constexpr int WIH_U16 = NT * KTX * 512;
  constexpr int WHH_U16 = NT * KTH * 512;
  constexpr int HS = H + 2;        // padded h row stride (words)
  constexpr int GS = 3 * H + 2;    // padded gate row stride (words)
  constexpr int PW = 16 * HS + 2 * 16 * GS;  // per-wave LDS floats

  const int lane = threadIdx.x & 31;
  const int wv   = threadIdx.x >> 5;
  const int bt   = blockIdx.x * 2 + wv;
  const int m    = lane & 15;
  const int hi   = lane >> 4;

  extern __shared__ float lds[];
  unsigned short* wihL = (unsigned short*)lds;
  unsigned short* whhL = wihL + (STAGE_W ? WIH_U16 : 0);
  float* fbase = (float*)(whhL + (STAGE_W ? WHH_U16 : 0));
  float* hbuf  = fbase + (size_t)wv * PW;
  float* gi    = hbuf + 16 * HS;
  float* gh    = gi + 16 * GS;

  if (STAGE_W) {
    for (int i = threadIdx.x; i < WIH_U16 / 8; i += 64)
      ((v4u*)wihL)[i] = ((const v4u*)wih_frag)[i];
    for (int i = threadIdx.x; i < WHH_U16 / 8; i += 64)
      ((v4u*)whhL)[i] = ((const v4u*)whh_frag)[i];
  }
  for (int i = lane; i < 16 * H; i += 32) {
    int r = i / H, f = i - r * H;
    hbuf[r * HS + f] = 0.0f;
  }
  __syncthreads();

  v16bf ax[KTX];

  auto load_ax = [&](int t) {
    const unsigned short* xp = x_frag + ((size_t)bt * TX + t) * KTX * 512;
    if (!XCONST && t + 1 < T)
      __builtin_prefetch(x_frag + ((size_t)bt * TX + (t + 1)) * KTX * 512, 0, 0);
#pragma unroll
    for (int kt = 0; kt < KTX; ++kt)
      ax[kt] = *reinterpret_cast<const v16bf*>(xp + ((size_t)kt * 32 + lane) * 16);
  };

  auto compute_gi = [&]() {
#pragma clang loop unroll(disable)
    for (int nt = 0; nt < NT; nt += 2) {
      v8f acc0, acc1;
      float bv0 = bih[nt * 16 + m];
      float bv1 = bih[nt * 16 + 16 + m];
#pragma unroll
      for (int r = 0; r < 8; ++r) { acc0[r] = bv0; acc1[r] = bv1; }
#pragma unroll
      for (int kt = 0; kt < KTX; ++kt) {
        v16bf b0, b1;
        if (STAGE_W) {
          b0 = *reinterpret_cast<const v16bf*>(
              wihL + ((size_t)(nt * KTX + kt) * 32 + lane) * 16);
          b1 = *reinterpret_cast<const v16bf*>(
              wihL + ((size_t)((nt + 1) * KTX + kt) * 32 + lane) * 16);
        } else {
          b0 = *reinterpret_cast<const v16bf*>(
              wih_frag + ((size_t)(nt * KTX + kt) * 32 + lane) * 16);
          b1 = *reinterpret_cast<const v16bf*>(
              wih_frag + ((size_t)((nt + 1) * KTX + kt) * 32 + lane) * 16);
        }
        acc0 = __builtin_amdgcn_wmma_f32_16x16x32_bf16(
            false, ax[kt], false, b0, (short)0, acc0, false, false);
        acc1 = __builtin_amdgcn_wmma_f32_16x16x32_bf16(
            false, ax[kt], false, b1, (short)0, acc1, false, false);
      }
      int col = nt * 16 + m;
#pragma unroll
      for (int r = 0; r < 8; ++r) {
        gi[(size_t)(r + 8 * hi) * GS + col]      = acc0[r];
        gi[(size_t)(r + 8 * hi) * GS + col + 16] = acc1[r];
      }
    }
  };

  auto compute_gh = [&]() {
    v16bf ah[KTH];
#pragma unroll
    for (int kt = 0; kt < KTH; ++kt) {
      BFrag u;
#pragma unroll
      for (int j = 0; j < 16; ++j) {
        int k = (j & 7) + 8 * hi + 16 * (j >> 3) + 32 * kt;
        u.u[j] = f2bf(hbuf[(size_t)m * HS + k]);
      }
      ah[kt] = u.v;
    }
#pragma clang loop unroll(disable)
    for (int nt = 0; nt < NT; nt += 2) {
      v8f acc0, acc1;
      float bv0 = bhh[nt * 16 + m];
      float bv1 = bhh[nt * 16 + 16 + m];
#pragma unroll
      for (int r = 0; r < 8; ++r) { acc0[r] = bv0; acc1[r] = bv1; }
#pragma unroll
      for (int kt = 0; kt < KTH; ++kt) {
        v16bf b0, b1;
        if (STAGE_W) {
          b0 = *reinterpret_cast<const v16bf*>(
              whhL + ((size_t)(nt * KTH + kt) * 32 + lane) * 16);
          b1 = *reinterpret_cast<const v16bf*>(
              whhL + ((size_t)((nt + 1) * KTH + kt) * 32 + lane) * 16);
        } else {
          b0 = *reinterpret_cast<const v16bf*>(
              whh_frag + ((size_t)(nt * KTH + kt) * 32 + lane) * 16);
          b1 = *reinterpret_cast<const v16bf*>(
              whh_frag + ((size_t)((nt + 1) * KTH + kt) * 32 + lane) * 16);
        }
        acc0 = __builtin_amdgcn_wmma_f32_16x16x32_bf16(
            false, ah[kt], false, b0, (short)0, acc0, false, false);
        acc1 = __builtin_amdgcn_wmma_f32_16x16x32_bf16(
            false, ah[kt], false, b1, (short)0, acc1, false, false);
      }
      int col = nt * 16 + m;
#pragma unroll
      for (int r = 0; r < 8; ++r) {
        gh[(size_t)(r + 8 * hi) * GS + col]      = acc0[r];
        gh[(size_t)(r + 8 * hi) * GS + col + 16] = acc1[r];
      }
    }
  };

  if (XCONST) { load_ax(0); compute_gi(); }

  for (int t = 0; t < T; ++t) {
    if (!XCONST) { load_ax(t); compute_gi(); }
    compute_gh();
    __syncthreads();
    for (int idx = lane; idx < 16 * H; idx += 32) {
      int r = idx / H, f = idx - r * H;
      float ir  = gi[(size_t)r * GS + f];
      float iz  = gi[(size_t)r * GS + H + f];
      float in_ = gi[(size_t)r * GS + 2 * H + f];
      float hr  = gh[(size_t)r * GS + f];
      float hz  = gh[(size_t)r * GS + H + f];
      float hn  = gh[(size_t)r * GS + 2 * H + f];
      float rg = 1.0f / (1.0f + __expf(-(ir + hr)));
      float zg = 1.0f / (1.0f + __expf(-(iz + hz)));
      float ng = tanhf(in_ + rg * hn);
      hbuf[(size_t)r * HS + f] = (1.0f - zg) * ng + zg * hbuf[(size_t)r * HS + f];
    }
    __syncthreads();
    if (HEADOUT) {
      int rr = lane >> 1, o = lane & 1;
      float s = head_b[o];
      for (int f = 0; f < H; ++f) s += hbuf[(size_t)rr * HS + f] * head_w[o * H + f];
      pred[(((size_t)bt * 16 + rr) * T + t) * 2 + o] = s;
      __syncthreads();
    }
  }

  if (!HEADOUT) {
    for (int idx = lane; idx < 16 * H; idx += 32) {
      int r = idx / H, f = idx - r * H;
      h_out[((size_t)bt * 16 + r) * H + f] = hbuf[(size_t)r * HS + f];
    }
  }
}

// ---------------------------------------------------------------------------
// Host launch
// ---------------------------------------------------------------------------
extern "C" void kernel_launch(void* const* d_in, const int* in_sizes, int n_in,
                              void* d_out, int out_size, void* d_ws, size_t ws_size,
                              hipStream_t stream) {
  (void)in_sizes; (void)n_in; (void)out_size; (void)ws_size;
  auto P = [&](int i) { return (const float*)d_in[i]; };

  const float* x_traj = P(0);
  const float* x_cf   = P(1);
  const float* x_car  = P(2);
  const float* y_traj = P(3);
  const float* eps    = P(4);

  // params flattened in JAX pytree (alphabetical) order
  const float *car_bhh = P(5),  *car_bih = P(6),  *car_whh = P(7),  *car_wih = P(8);
  const float *cf_bhh  = P(9),  *cf_bih  = P(10), *cf_whh  = P(11), *cf_wih  = P(12);
  const float *dec_bhh = P(13), *dec_bih = P(14), *dec_whh = P(15), *dec_wih = P(16);
  const float *declin_b = P(17), *declin_w = P(18);
  const float *embcar_b = P(19), *embcar_w = P(20);
  const float *embcf_b  = P(21), *embcf_w  = P(22);
  const float *embtraj_b = P(23), *embtraj_w = P(24);
  const float *head_b = P(25), *head_w = P(26);
  const float *m_ac_ib = P(27), *m_ac_iw = P(28), *m_ac_ob = P(29), *m_ac_ow = P(30); // car_x_cf
  const float *m_at_ib = P(31), *m_at_iw = P(32), *m_at_ob = P(33), *m_at_ow = P(34); // car_x_traj
  const float *m_ca_ib = P(35), *m_ca_iw = P(36), *m_ca_ob = P(37), *m_ca_ow = P(38); // cf_x_car
  const float *m_ct_ib = P(39), *m_ct_iw = P(40), *m_ct_ob = P(41), *m_ct_ow = P(42); // cf_x_traj
  const float *m_ta_ib = P(43), *m_ta_iw = P(44), *m_ta_ob = P(45), *m_ta_ow = P(46); // traj_x_car
  const float *m_tc_ib = P(47), *m_tc_iw = P(48), *m_tc_ob = P(49), *m_tc_ow = P(50); // traj_x_cf
  const float *mu_b = P(51), *mu_w = P(52);
  const float *traj_bhh = P(53), *traj_bih = P(54), *traj_whh = P(55), *traj_wih = P(56);
  const float *var_b = P(57), *var_w = P(58);
  const float *y_bhh = P(59), *y_bih = P(60), *y_whh = P(61), *y_wih = P(62);
  const float *ylin_b = P(63), *ylin_w = P(64);

  // output layout: pred [B,80,2] | mean [B,32] | log_var [B,32]
  float* out      = (float*)d_out;
  float* pred     = out;
  float* mean_out = out + (size_t)B_SZ * T_PRED * 2;
  float* lv_out   = mean_out + (size_t)B_SZ * 32;

  // workspace bump allocator
  char* wsp = (char*)d_ws;
  auto alloc = [&](size_t bytes) -> void* {
    void* p = (void*)wsp;
    wsp += (bytes + 255) & ~(size_t)255;
    return p;
  };
  const int BT = B_SZ / 16;
  unsigned short* ef_t = (unsigned short*)alloc((size_t)RT_OBS * 512 * 2);
  unsigned short* ef_c = (unsigned short*)alloc((size_t)RT_OBS * 512 * 2);
  unsigned short* ef_a = (unsigned short*)alloc((size_t)RT_OBS * 512 * 2);
  float* qkv = (float*)alloc((size_t)B_SZ * T_OBS * 96 * 4);
  unsigned short* o_frag = (unsigned short*)alloc((size_t)RT_OBS * 512 * 2);
  unsigned short* s_traj = (unsigned short*)alloc((size_t)BT * T_OBS * 2 * 512 * 2);
  unsigned short* s_cf   = (unsigned short*)alloc((size_t)BT * T_OBS * 2 * 512 * 2);
  unsigned short* s_car  = (unsigned short*)alloc((size_t)BT * T_OBS * 2 * 512 * 2);
  unsigned short* y_frag = (unsigned short*)alloc((size_t)BT * T_PRED * 4 * 512 * 2);
  unsigned short* dl_frag = (unsigned short*)alloc((size_t)BT * 4 * 512 * 2);
  float* h_traj = (float*)alloc((size_t)B_SZ * 128 * 4);
  float* h_cf   = (float*)alloc((size_t)B_SZ * 128 * 4);
  float* h_car  = (float*)alloc((size_t)B_SZ * 128 * 4);
  float* h_y    = (float*)alloc((size_t)B_SZ * 256 * 4);
  unsigned short* wT_ih = (unsigned short*)alloc((size_t)24 * 2 * 512 * 2);
  unsigned short* wT_hh = (unsigned short*)alloc((size_t)24 * 4 * 512 * 2);
  unsigned short* wC_ih = (unsigned short*)alloc((size_t)24 * 2 * 512 * 2);
  unsigned short* wC_hh = (unsigned short*)alloc((size_t)24 * 4 * 512 * 2);
  unsigned short* wA_ih = (unsigned short*)alloc((size_t)24 * 2 * 512 * 2);
  unsigned short* wA_hh = (unsigned short*)alloc((size_t)24 * 4 * 512 * 2);
  unsigned short* wY_ih = (unsigned short*)alloc((size_t)48 * 4 * 512 * 2);
  unsigned short* wY_hh = (unsigned short*)alloc((size_t)48 * 8 * 512 * 2);
  unsigned short* wD_ih = (unsigned short*)alloc((size_t)24 * 4 * 512 * 2);
  unsigned short* wD_hh = (unsigned short*)alloc((size_t)24 * 4 * 512 * 2);
  unsigned short* miwF[6];
  unsigned short* mowF[6];
  for (int i = 0; i < 6; ++i) {
    miwF[i] = (unsigned short*)alloc((size_t)6 * 512 * 2);
    mowF[i] = (unsigned short*)alloc((size_t)2 * 512 * 2);
  }

  auto pack = [&](const float* W, int N, int K, unsigned short* dst) {
    int total = (N / 16) * (K / 32) * 512;
    pack_w_kernel<<<(total + 255) / 256, 256, 0, stream>>>(W, N, K, dst);
  };
  pack(traj_wih, 384, 64, wT_ih);  pack(traj_whh, 384, 128, wT_hh);
  pack(cf_wih,   384, 64, wC_ih);  pack(cf_whh,   384, 128, wC_hh);
  pack(car_wih,  384, 64, wA_ih);  pack(car_whh,  384, 128, wA_hh);
  pack(y_wih,    768, 128, wY_ih); pack(y_whh,    768, 256, wY_hh);
  pack(dec_wih,  384, 128, wD_ih); pack(dec_whh,  384, 128, wD_hh);

  // MHA order: tc, ta, ct, ca, at, ac
  const float* in_ws[6]  = {m_tc_iw, m_ta_iw, m_ct_iw, m_ca_iw, m_at_iw, m_ac_iw};
  const float* in_bs[6]  = {m_tc_ib, m_ta_ib, m_ct_ib, m_ca_ib, m_at_ib, m_ac_ib};
  const float* out_ws[6] = {m_tc_ow, m_ta_ow, m_ct_ow, m_ca_ow, m_at_ow, m_ac_ow};
  const float* out_bs[6] = {m_tc_ob, m_ta_ob, m_ct_ob, m_ca_ob, m_at_ob, m_ac_ob};
  for (int i = 0; i < 6; ++i) {
    pack(in_ws[i], 96, 32, miwF[i]);
    pack(out_ws[i], 32, 32, mowF[i]);
  }

  // embeddings -> token-major fragments
  {
    int total = B_SZ * T_OBS * 32;
    emb_frag_kernel<<<(total + 255) / 256, 256, 0, stream>>>(x_traj, embtraj_w, embtraj_b, ef_t, total);
    emb_frag_kernel<<<(total + 255) / 256, 256, 0, stream>>>(x_cf,   embcf_w,   embcf_b,   ef_c, total);
    emb_frag_kernel<<<(total + 255) / 256, 256, 0, stream>>>(x_car,  embcar_w,  embcar_b,  ef_a, total);
  }

  // cross attentions: WMMA QKV -> VALU attention core -> WMMA out-projection
  // s_cf = [ac|tc], s_car = [ca|ta], s_traj = [ct|at]
  const unsigned short* qsrcs[6]  = {ef_t, ef_t, ef_c, ef_c, ef_a, ef_a};
  const unsigned short* kvsrcs[6] = {ef_c, ef_a, ef_t, ef_a, ef_t, ef_c};
  unsigned short* dsts[6] = {s_cf, s_car, s_traj, s_car, s_traj, s_cf};
  const int kts[6] = {1, 1, 0, 0, 1, 0};
  const size_t attn_lds = (size_t)(1920 * 3 + 3600 + 1920) * 4;
  for (int i = 0; i < 6; ++i) {
    qkv_gemm_kernel<<<RT_OBS / 8, 256, 0, stream>>>(qsrcs[i], kvsrcs[i], miwF[i], in_bs[i], qkv);
    attn_kernel<<<B_SZ, 64, attn_lds, stream>>>(qkv, o_frag);
    outproj_gemm_kernel<<<RT_OBS / 8, 256, 0, stream>>>(o_frag, mowF[i], out_bs[i], dsts[i], kts[i]);
  }

  // y embedding
  {
    int total = B_SZ * T_PRED * 128;
    ylin_kernel<<<(total + 255) / 256, 256, 0, stream>>>(y_traj, ylin_w, ylin_b, y_frag, total);
  }

  // per-wave padded LDS floats: 16*(H+2) + 2*16*(3H+2)
  auto pw_bytes = [](int H) {
    return (size_t)(16 * (H + 2) + 2 * 16 * (3 * H + 2)) * 4;
  };

  // encoder GRUs: 2 waves/WG share LDS-staged weights.
  const size_t lds_enc = (size_t)(24*2*512 + 24*4*512) * 2 + 2 * pw_bytes(128);
  gru_wmma_kernel<64, 128, T_OBS, false, false, true><<<BT/2, 64, lds_enc, stream>>>(
      s_traj, wT_ih, wT_hh, traj_bih, traj_bhh, h_traj, nullptr, nullptr, nullptr);
  gru_wmma_kernel<64, 128, T_OBS, false, false, true><<<BT/2, 64, lds_enc, stream>>>(
      s_cf, wC_ih, wC_hh, cf_bih, cf_bhh, h_cf, nullptr, nullptr, nullptr);
  gru_wmma_kernel<64, 128, T_OBS, false, false, true><<<BT/2, 64, lds_enc, stream>>>(
      s_car, wA_ih, wA_hh, car_bih, car_bhh, h_car, nullptr, nullptr, nullptr);

  // posterior GRU: packed weights (590 KB) exceed LDS -> global/L2.
  const size_t lds_y = 2 * pw_bytes(256);
  gru_wmma_kernel<128, 256, T_PRED, false, false, false><<<BT/2, 64, lds_y, stream>>>(
      y_frag, wY_ih, wY_hh, y_bih, y_bhh, h_y, nullptr, nullptr, nullptr);

  // latent + decoder input
  latent_kernel<<<B_SZ, 64, 0, stream>>>(h_traj, h_cf, h_car, h_y, eps,
                                         mu_w, mu_b, var_w, var_b,
                                         declin_w, declin_b,
                                         mean_out, lv_out, dl_frag);

  // decoder GRU with per-step head (constant input over time).
  const size_t lds_dec = (size_t)(24*4*512 + 24*4*512) * 2 + 2 * pw_bytes(128);
  gru_wmma_kernel<128, 128, T_PRED, true, true, true><<<BT/2, 64, lds_dec, stream>>>(
      dl_frag, wD_ih, wD_hh, dec_bih, dec_bhh, nullptr, head_w, head_b, pred);
}
